// LRNN_StatefulFFN_22703197127368
// MI455X (gfx1250) — compile-verified, matrix operand order checked by reading, not run on hardware
//
#include <hip/hip_runtime.h>
#include <math.h>

// ---------------- problem constants ----------------
#define BATCH    2
#define SEQ      1024
#define DMODEL   512
#define DEXPAND  1024
#define DSTATE   16
#define M_TOT    (BATCH * SEQ)        // 2048
#define N1       (2 * DEXPAND)        // 2048 (in-proj output)
#define K1       DMODEL               // 512
#define K2       DEXPAND              // 1024
#define N2       DMODEL               // 512
#define PCOLS    (4 * DSTATE + 1)     // 65

typedef __attribute__((ext_vector_type(16))) __bf16 v16bf;
typedef __attribute__((ext_vector_type(8)))  __bf16 v8bf;
typedef __attribute__((ext_vector_type(8)))  float  v8f;
typedef __attribute__((ext_vector_type(4)))  float  v4f;
typedef __attribute__((ext_vector_type(4)))  int    v4i;

// ---------------- gfx1250 async global->LDS availability ----------------
#if defined(__has_builtin)
#if __has_builtin(__builtin_amdgcn_global_load_async_to_lds_b128)
#define ASYNC_LDS 1
#endif
#endif
#ifndef ASYNC_LDS
#define ASYNC_LDS 0
#endif

typedef __attribute__((address_space(1))) v4i* gv4i_t;   // global int4*
typedef __attribute__((address_space(3))) v4i* lv4i_t;   // LDS int4*

__device__ __forceinline__ void async_fence_barrier() {
#if ASYNC_LDS
#if __has_builtin(__builtin_amdgcn_s_wait_asynccnt)
  __builtin_amdgcn_s_wait_asynccnt(0);
#else
  asm volatile("s_wait_asynccnt 0x0" ::: "memory");
#endif
#endif
  __syncthreads();
}

// ---------------- f32 -> bf16 convert ----------------
__global__ void cvt_bf16_kernel(const float* __restrict__ src,
                                unsigned short* __restrict__ dst, int n) {
  __bf16* d = reinterpret_cast<__bf16*>(dst);
  for (int i = blockIdx.x * blockDim.x + threadIdx.x; i < n;
       i += gridDim.x * blockDim.x) {
    d[i] = (__bf16)src[i];
  }
}

// ---------------- LDS-staged, double-buffered WMMA GEMM core ---------------
// Block = 256 threads (8 waves). Block tile: M=128, N=64, Kstep=32.
// LDS: A tile 128x32 bf16 + B tile 64x32 bf16, rows padded to 40 elems (80B).
#define LDS_ROWPAD 40
#define LDSA_ELEMS (128 * LDS_ROWPAD)            // 5120
#define LDSB_ELEMS (64 * LDS_ROWPAD)             // 2560
#define BUF_ELEMS  (LDSA_ELEMS + LDSB_ELEMS)     // 7680

__device__ __forceinline__ void stage_tile(const __bf16* __restrict__ A,
                                           const __bf16* __restrict__ W,
                                           __bf16* shm, int buf, int m0, int n0,
                                           int kk, int K, int tid) {
  // 192 rows x 4 chunks of 16B = 768 chunks; 256 threads x 3.
#pragma unroll
  for (int i = 0; i < 3; ++i) {
    const int c   = tid + i * 256;
    const int row = c >> 2;
    const int cic = c & 3;
    const int ldsIdx =
        buf * BUF_ELEMS +
        (row < 128 ? row * LDS_ROWPAD : LDSA_ELEMS + (row - 128) * LDS_ROWPAD) +
        cic * 8;
    const __bf16* gsrc =
        (row < 128) ? (A + (size_t)(m0 + row) * K + kk + cic * 8)
                    : (W + (size_t)(n0 + row - 128) * K + kk + cic * 8);
#if ASYNC_LDS
    __builtin_amdgcn_global_load_async_to_lds_b128(
        (gv4i_t)gsrc, (lv4i_t)(&shm[ldsIdx]), 0, 0);
#else
    *reinterpret_cast<v8bf*>(&shm[ldsIdx]) =
        *reinterpret_cast<const v8bf*>(gsrc);
#endif
  }
}

// Computes 4x v8f accumulators (16M x 64N per wave) over K.
__device__ __forceinline__ void gemm_tile_body(const __bf16* __restrict__ A,
                                               const __bf16* __restrict__ W,
                                               __bf16* shm, int m0, int n0,
                                               int K, v8f acc[4]) {
  const int tid  = threadIdx.x;
  const int lane = tid & 31;
  const int wave = tid >> 5;
  const int half = lane >> 4;
  const int l16  = lane & 15;
  const int kOfs = half * 8;      // per-lane K base inside the 32-chunk

#pragma unroll
  for (int j = 0; j < 4; ++j)
#pragma unroll
    for (int r = 0; r < 8; ++r) acc[j][r] = 0.0f;

  stage_tile(A, W, shm, 0, m0, n0, 0, K, tid);
  async_fence_barrier();

  const int nsteps = K / 32;
  for (int step = 0; step < nsteps; ++step) {
    const int cur = step & 1;
    if (step + 1 < nsteps)
      stage_tile(A, W, shm, 1 - cur, m0, n0, (step + 1) * 32, K, tid);

    const int aIdx = cur * BUF_ELEMS + (wave * 16 + l16) * LDS_ROWPAD + kOfs;
    v8bf alo = *reinterpret_cast<const v8bf*>(&shm[aIdx]);
    v8bf ahi = *reinterpret_cast<const v8bf*>(&shm[aIdx + 16]);
    v16bf afrag;
#pragma unroll
    for (int i = 0; i < 8; ++i) { afrag[i] = alo[i]; afrag[i + 8] = ahi[i]; }

#pragma unroll
    for (int j = 0; j < 4; ++j) {
      const int bIdx = cur * BUF_ELEMS + LDSA_ELEMS +
                       (j * 16 + l16) * LDS_ROWPAD + kOfs;
      v8bf blo = *reinterpret_cast<const v8bf*>(&shm[bIdx]);
      v8bf bhi = *reinterpret_cast<const v8bf*>(&shm[bIdx + 16]);
      v16bf bfrag;
#pragma unroll
      for (int i = 0; i < 8; ++i) { bfrag[i] = blo[i]; bfrag[i + 8] = bhi[i]; }
      acc[j] = __builtin_amdgcn_wmma_f32_16x16x32_bf16(
          false, afrag, false, bfrag, (short)0, acc[j], false, false);
    }
    async_fence_barrier();
  }
}

// ---------------- GEMM1: proj = x @ in_w^T + in_b; silu/split epilogue ------
__global__ void __launch_bounds__(256)
gemm_inproj_kernel(const unsigned short* __restrict__ Aus,
                   const unsigned short* __restrict__ Wus,
                   const float* __restrict__ bias, float* __restrict__ gate,
                   float* __restrict__ xssm) {
  __shared__ __align__(16) __bf16 shm[2 * BUF_ELEMS];
  const __bf16* A = reinterpret_cast<const __bf16*>(Aus);
  const __bf16* W = reinterpret_cast<const __bf16*>(Wus);
  const int lane = threadIdx.x & 31;
  const int wave = threadIdx.x >> 5;
  const int half = lane >> 4;
  const int l16  = lane & 15;
  const int mb = blockIdx.x >> 5;   // 16 M-blocks
  const int nb = blockIdx.x & 31;   // 32 N-blocks
  const int m0 = mb * 128;
  const int n0 = nb * 64;

  v8f acc[4];
  gemm_tile_body(A, W, shm, m0, n0, K1, acc);

  const int mbase = m0 + wave * 16 + (half ? 8 : 0);
#pragma unroll
  for (int j = 0; j < 4; ++j) {
    const int n = n0 + j * 16 + l16;
    const float bn = bias[n];
#pragma unroll
    for (int r = 0; r < 8; ++r) {
      const int m = mbase + r;
      const float v = acc[j][r] + bn;
      if (n < DEXPAND) {
        gate[(size_t)m * DEXPAND + n] = v / (1.0f + __expf(-v));  // silu
      } else {
        xssm[(size_t)m * DEXPAND + (n - DEXPAND)] = v;
      }
    }
  }
}

// ---------------- GEMM2: out = y2 @ out_w^T + out_b -------------------------
__global__ void __launch_bounds__(256)
gemm_out_kernel(const unsigned short* __restrict__ Aus,
                const unsigned short* __restrict__ Wus,
                const float* __restrict__ bias, float* __restrict__ out) {
  __shared__ __align__(16) __bf16 shm[2 * BUF_ELEMS];
  const __bf16* A = reinterpret_cast<const __bf16*>(Aus);
  const __bf16* W = reinterpret_cast<const __bf16*>(Wus);
  const int lane = threadIdx.x & 31;
  const int wave = threadIdx.x >> 5;
  const int half = lane >> 4;
  const int l16  = lane & 15;
  const int mb = blockIdx.x >> 3;   // 16 M-blocks
  const int nb = blockIdx.x & 7;    // 8 N-blocks
  const int m0 = mb * 128;
  const int n0 = nb * 64;

  v8f acc[4];
  gemm_tile_body(A, W, shm, m0, n0, K2, acc);

  const int mbase = m0 + wave * 16 + (half ? 8 : 0);
#pragma unroll
  for (int j = 0; j < 4; ++j) {
    const int n = n0 + j * 16 + l16;
    const float bn = bias[n];
#pragma unroll
    for (int r = 0; r < 8; ++r) {
      const int m = mbase + r;
      out[(size_t)m * N2 + n] = acc[j][r] + bn;
    }
  }
}

// ---------------- ssm projection: p = x_ssm @ ssm_w^T + ssm_b (N=65) -------
__global__ void ssm_proj_kernel(const float* __restrict__ xssm,
                                const float* __restrict__ sw,
                                const float* __restrict__ sb,
                                float* __restrict__ p) {
  const int idx = blockIdx.x * blockDim.x + threadIdx.x;
  if (idx >= M_TOT * PCOLS) return;
  const int m = idx / PCOLS;
  const int j = idx % PCOLS;
  const v4f* xr = reinterpret_cast<const v4f*>(xssm + (size_t)m * K2);
  const v4f* wr = reinterpret_cast<const v4f*>(sw + (size_t)j * K2);
  float acc = 0.0f;
  for (int k = 0; k < K2 / 4; ++k) {
    v4f a = xr[k], b = wr[k];
    acc += a[0] * b[0] + a[1] * b[1] + a[2] * b[2] + a[3] * b[3];
  }
  p[idx] = acc + sb[j];
}

// ---------------- sequential complex diagonal scan --------------------------
__global__ void scan_kernel(const float* __restrict__ p,
                            const float* __restrict__ xssm,
                            const float* __restrict__ Are,
                            const float* __restrict__ Aim,
                            float* __restrict__ y) {
  const int lane = threadIdx.x & 31;
  const int wave = threadIdx.x >> 5;
  const int half = lane >> 4;
  const int s    = lane & 15;
  const int row  = blockIdx.x * 16 + wave * 2 + half;   // 0..2047
  const int b    = row >> 10;
  const int e    = row & 1023;

  const float a_re = Are[e * DSTATE + s];
  const float a_im = Aim[e * DSTATE + s];
  const float eps  = 1.1920929e-7f;

  float h_re = 0.0f, h_im = 0.0f;
  for (int t = 0; t < SEQ; ++t) {
    const int m = b * SEQ + t;
    const float* pr = p + (size_t)m * PCOLS;
    const float draw = pr[4 * DSTATE];
    const float delta = (draw > 20.0f) ? draw : log1pf(__expf(draw));
    const float xv  = xssm[(size_t)m * DEXPAND + e];
    const float Cre = pr[2 * DSTATE + s];
    const float Cim = pr[3 * DSTATE + s];

    const float dAr = delta * a_re;
    const float dAi = delta * a_im;
    const float er  = __expf(dAr);
    const float Abr = er * __cosf(dAi);
    const float Abi = er * __sinf(dAi);

    float Bbr, Bbi;
    const float mag2 = dAr * dAr + dAi * dAi;
    if (mag2 < 1e-6f) {
      Bbr = delta * (1.0f + 0.5f * dAr);
      Bbi = delta * (0.5f * dAi);
    } else {
      const float dr = dAr + eps, di = dAi;
      const float inv = 1.0f / (dr * dr + di * di);
      const float nr = Abr - 1.0f, ni = Abi;
      Bbr = delta * (nr * dr + ni * di) * inv;
      Bbi = delta * (ni * dr - nr * di) * inv;
    }

    const float hr = h_re, hi = h_im;
    h_re = Abr * hr - Abi * hi + Bbr * xv;
    h_im = Abr * hi + Abi * hr + Bbi * xv;

    float yc = h_re * Cre + h_im * Cim;   // Re(h * conj(C))
#pragma unroll
    for (int off = 8; off > 0; off >>= 1)
      yc += __shfl_xor(yc, off, 16);
    if (s == 0) y[(size_t)m * DEXPAND + e] = yc;
  }
}

// ---------------- elementwise: y2 = (y + x_ssm*D) * gate -> bf16 ------------
__global__ void mix_kernel(const float* __restrict__ y,
                           const float* __restrict__ xssm,
                           const float* __restrict__ gate,
                           const float* __restrict__ D,
                           unsigned short* __restrict__ y2us) {
  __bf16* y2 = reinterpret_cast<__bf16*>(y2us);
  for (int i = blockIdx.x * blockDim.x + threadIdx.x; i < M_TOT * DEXPAND;
       i += gridDim.x * blockDim.x) {
    const int k = i & (DEXPAND - 1);
    const float v = (y[i] + xssm[i] * D[k]) * gate[i];
    y2[i] = (__bf16)v;
  }
}

// ---------------- host launch ----------------
extern "C" void kernel_launch(void* const* d_in, const int* in_sizes, int n_in,
                              void* d_out, int out_size, void* d_ws, size_t ws_size,
                              hipStream_t stream) {
  const float* x      = (const float*)d_in[0];
  const float* in_w   = (const float*)d_in[1];
  const float* in_b   = (const float*)d_in[2];
  const float* ssm_w  = (const float*)d_in[3];
  const float* ssm_b  = (const float*)d_in[4];
  const float* out_w  = (const float*)d_in[5];
  const float* out_b  = (const float*)d_in[6];
  const float* A_re   = (const float*)d_in[7];
  const float* A_im   = (const float*)d_in[8];
  const float* Dvec   = (const float*)d_in[9];
  float* out = (float*)d_out;

  // workspace layout (bytes)
  char* ws = (char*)d_ws;
  unsigned short* x_bf    = (unsigned short*)(ws);                       // 2 MB
  unsigned short* inw_bf  = (unsigned short*)(ws + (size_t)2097152);     // 2 MB
  unsigned short* outw_bf = (unsigned short*)(ws + (size_t)4194304);     // 1 MB
  float* gate             = (float*)(ws + (size_t)5242880);              // 8 MB
  float* xssm             = (float*)(ws + (size_t)13631488);             // 8 MB
  float* p                = (float*)(ws + (size_t)22020096);             // 0.52 MB
  float* ybuf             = (float*)(ws + (size_t)22552576);             // 8 MB
  unsigned short* y2_bf   = (unsigned short*)(ws + (size_t)30941184);    // 4 MB

  // 1) bf16 staging of GEMM operands
  cvt_bf16_kernel<<<2048, 256, 0, stream>>>(x, x_bf, M_TOT * K1);
  cvt_bf16_kernel<<<2048, 256, 0, stream>>>(in_w, inw_bf, N1 * K1);
  cvt_bf16_kernel<<<1024, 256, 0, stream>>>(out_w, outw_bf, N2 * K2);

  // 2) in-projection GEMM (WMMA, LDS double-buffered) + silu/split epilogue
  gemm_inproj_kernel<<<512, 256, 0, stream>>>(x_bf, inw_bf, in_b, gate, xssm);

  // 3) ssm projection (skinny N=65)
  ssm_proj_kernel<<<(M_TOT * PCOLS + 255) / 256, 256, 0, stream>>>(xssm, ssm_w,
                                                                   ssm_b, p);

  // 4) sequential complex scan
  scan_kernel<<<128, 256, 0, stream>>>(p, xssm, A_re, A_im, ybuf);

  // 5) y2 = (y + x_ssm*D) * gate  -> bf16
  mix_kernel<<<4096, 256, 0, stream>>>(ybuf, xssm, gate, Dvec, y2_bf);

  // 6) out-projection GEMM (WMMA, LDS double-buffered) + bias
  gemm_out_kernel<<<128, 256, 0, stream>>>(y2_bf, outw_bf, out_b, out);
}
